// KroneckerSparseLinear_80178449481863
// MI455X (gfx1250) — matrix operand order repaired and checked by reading, not compiled, not run on hardware
//
#include <hip/hip_runtime.h>
#include <hip/hip_bf16.h>

// ---------------------------------------------------------------------------
// KroneckerSparseLinear for MI455X (gfx1250, wave32, WMMA)
//
//   Stage 1: W^T (bf16 hi/lo split) = KronPermute(A @ B) + Sparse   [tiny]
//   Stage 2: out = X @ W + bias as bf16x3 WMMA GEMM (f32 accumulate),
//            double-buffered LDS, W tiles via async global->LDS copies.
// ---------------------------------------------------------------------------

typedef __attribute__((ext_vector_type(16))) __bf16 v16bf;
typedef __attribute__((ext_vector_type(8)))  float  v8f;
typedef int vi4 __attribute__((vector_size(16)));   // gcc-style int4 (builtin ABI)
typedef unsigned short u16;
typedef unsigned int   u32;

#define AS1 __attribute__((address_space(1)))
#define AS3 __attribute__((address_space(3)))

#define GEMM_M 16384   // 4 * 4096
#define GEMM_K 768
#define GEMM_N 3072
#define KC     32      // K chunk (bf16 WMMA depth)
#define NKC    (GEMM_K / KC)   // 24 (even -> clean 2x unrolled ping-pong)
#define LDSP   40      // padded row length in bf16 units (32 + 8) = 80 bytes

__device__ __forceinline__ u16 f32_to_bf16_rne(float f) {
  u32 u = __float_as_uint(f);
  u32 r = 0x7FFFu + ((u >> 16) & 1u);
  return (u16)((u + r) >> 16);
}
__device__ __forceinline__ float bf16_to_f32(u16 h) {
  return __uint_as_float(((u32)h) << 16);
}

// 16-byte async copy global -> LDS (per-lane), tracked by ASYNCcnt.
__device__ __forceinline__ void async_cp16(const void* gsrc, void* lds_dst) {
#if __has_builtin(__builtin_amdgcn_global_load_async_to_lds_b128)
  __builtin_amdgcn_global_load_async_to_lds_b128(
      (AS1 vi4*)gsrc, (AS3 vi4*)lds_dst, 0, 0);
#else
  unsigned lds_off = (unsigned)(unsigned long long)lds_dst;
  asm volatile("global_load_async_to_lds_b128 %0, %1, off"
               :
               : "v"(lds_off), "v"(gsrc)
               : "memory");
#endif
}

__device__ __forceinline__ void wait_async0() {
#if __has_builtin(__builtin_amdgcn_s_wait_asynccnt)
  __builtin_amdgcn_s_wait_asynccnt(0);
#else
  asm volatile("s_wait_asynccnt 0x0" ::: "memory");
#endif
}

// ---------------------------------------------------------------------------
// Stage 1: build W^T as bf16 hi/lo.
//   W[r][c], r in [0,768): m1 = r/24, m2 = r%24
//            c in [0,3072): n1 = c/48, n2 = c%48
//   W[r][c] = dot(A[m1*64+n1, :64], B[:64, m2*48+n2]) + Sparse[r][c]
//   Stored transposed: WT[c*768 + r] so the GEMM B-operand is K-contiguous.
// ---------------------------------------------------------------------------
__global__ __launch_bounds__(256)
void build_wt_kernel(const float* __restrict__ Af,   // [2048, 64]
                     const float* __restrict__ Bf,   // [64, 1152]
                     const float* __restrict__ Sp,   // [768, 3072]
                     u16* __restrict__ wt_hi,        // [3072, 768]
                     u16* __restrict__ wt_lo) {
  int t = blockIdx.x * 256 + threadIdx.x;    // 0 .. 768*3072-1
  int c = t / GEMM_K;
  int r = t - c * GEMM_K;
  int m1 = r / 24, m2 = r - m1 * 24;
  int n1 = c / 48, n2 = c - n1 * 48;
  const float* arow = Af + (size_t)(m1 * 64 + n1) * 64;
  const float* bcol = Bf + (m2 * 48 + n2);
  float acc = Sp[(size_t)r * GEMM_N + c];
#pragma unroll
  for (int j = 0; j < 64; ++j)
    acc = fmaf(arow[j], bcol[(size_t)j * 1152], acc);
  u16 h = f32_to_bf16_rne(acc);
  u16 l = f32_to_bf16_rne(acc - bf16_to_f32(h));
  wt_hi[(size_t)c * GEMM_K + r] = h;
  wt_lo[(size_t)c * GEMM_K + r] = l;
}

// ---------------------------------------------------------------------------
// Stage 2: out[16384,3072] = X[16384,768] @ W + bias, bf16x3 via WMMA.
// Block: 128x128 tile, 256 threads = 8 waves (4 in M x 2 in N).
// Wave: 32x64 patch = 2x4 tiles of 16x16, 8 x v8f accumulators.
// Double-buffered LDS; W via async-to-LDS, X via load+convert+ds_store.
// ---------------------------------------------------------------------------
__global__ __launch_bounds__(256)
void kron_gemm_kernel(const float* __restrict__ X,     // [16384, 768] f32
                      const u16*  __restrict__ wt_hi,  // [3072, 768] bf16
                      const u16*  __restrict__ wt_lo,  // [3072, 768] bf16
                      const float* __restrict__ bias,  // [3072]
                      float* __restrict__ out) {       // [16384, 3072]
  __shared__ __align__(16) u16 sA_hi[2][128 * LDSP];
  __shared__ __align__(16) u16 sA_lo[2][128 * LDSP];
  __shared__ __align__(16) u16 sB_hi[2][128 * LDSP];
  __shared__ __align__(16) u16 sB_lo[2][128 * LDSP];

  const int tid    = threadIdx.x;
  const int lane   = tid & 31;
  const int wid    = tid >> 5;
  const int wave_m = wid & 3;    // 0..3  -> 32-row band
  const int wave_n = wid >> 2;   // 0..1  -> 64-col band
  const int l16    = lane & 15;
  const int lhalf  = lane >> 4;  // 0 or 1

  const int bm0 = blockIdx.y * 128;   // row base in M
  const int bn0 = blockIdx.x * 128;   // col base in N

  const v8f zacc = {0.f, 0.f, 0.f, 0.f, 0.f, 0.f, 0.f, 0.f};
  v8f acc[2][4];
#pragma unroll
  for (int mi = 0; mi < 2; ++mi)
#pragma unroll
    for (int ni = 0; ni < 4; ++ni) acc[mi][ni] = zacc;

  union Frag { uint4 u[2]; v16bf v; };

  const float* Xb  = X     + (size_t)bm0 * GEMM_K;
  const u16*   Whb = wt_hi + (size_t)bn0 * GEMM_K;
  const u16*   Wlb = wt_lo + (size_t)bn0 * GEMM_K;

  // ---- stagers -------------------------------------------------------------
  auto load_w_async = [&](int kc, int buf) {
    const int k0 = kc * KC;
#pragma unroll
    for (int j = 0; j < 2; ++j) {
      int i   = tid + 256 * j;        // 0..511 16B slots
      int n   = i >> 2;               // 0..127
      int kch = i & 3;                // 0..3 (groups of 8 bf16)
      size_t go = (size_t)n * GEMM_K + k0 + kch * 8;
      int    lo = n * LDSP + kch * 8;
      async_cp16(Whb + go, &sB_hi[buf][lo]);
      async_cp16(Wlb + go, &sB_lo[buf][lo]);
    }
  };

  auto load_x = [&](int kc, int buf) {
    const int k0 = kc * KC;
#pragma unroll
    for (int j = 0; j < 4; ++j) {
      int i   = tid + 256 * j;        // 0..1023 float4 slots
      int row = i >> 3;               // 0..127
      int k4  = i & 7;                // 0..7 (groups of 4 floats)
      float4 xv = *(const float4*)(Xb + (size_t)row * GEMM_K + k0 + k4 * 4);
      u16 h0 = f32_to_bf16_rne(xv.x), h1 = f32_to_bf16_rne(xv.y);
      u16 h2 = f32_to_bf16_rne(xv.z), h3 = f32_to_bf16_rne(xv.w);
      u16 l0 = f32_to_bf16_rne(xv.x - bf16_to_f32(h0));
      u16 l1 = f32_to_bf16_rne(xv.y - bf16_to_f32(h1));
      u16 l2 = f32_to_bf16_rne(xv.z - bf16_to_f32(h2));
      u16 l3 = f32_to_bf16_rne(xv.w - bf16_to_f32(h3));
      uint2 hp = make_uint2((u32)h0 | ((u32)h1 << 16), (u32)h2 | ((u32)h3 << 16));
      uint2 lp = make_uint2((u32)l0 | ((u32)l1 << 16), (u32)l2 | ((u32)l3 << 16));
      *(uint2*)&sA_hi[buf][row * LDSP + k4 * 4] = hp;
      *(uint2*)&sA_lo[buf][row * LDSP + k4 * 4] = lp;
    }
  };

  // ---- compute: bf16x3 (hi*hi + hi*lo + lo*hi) -----------------------------
  auto compute = [&](int buf) {
#pragma unroll
    for (int mi = 0; mi < 2; ++mi) {
      // A operand 16x32 bf16: lane<16 holds row=l16, K {0..7, 16..23};
      //                       lane>=16 holds row=l16, K {8..15, 24..31}
      int arow = wave_m * 32 + mi * 16 + l16;
      int akb  = lhalf * 8;   // bf16 units
      Frag ah, al;
      ah.u[0] = *(const uint4*)&sA_hi[buf][arow * LDSP + akb];
      ah.u[1] = *(const uint4*)&sA_hi[buf][arow * LDSP + akb + 16];
      al.u[0] = *(const uint4*)&sA_lo[buf][arow * LDSP + akb];
      al.u[1] = *(const uint4*)&sA_lo[buf][arow * LDSP + akb + 16];
#pragma unroll
      for (int ni = 0; ni < 4; ++ni) {
        // B operand 32x16 bf16: lane<16 holds col=l16, K {0..7, 16..23};
        //                       lane>=16 holds col=l16, K {8..15, 24..31}
        int bcol = wave_n * 64 + ni * 16 + l16;
        int bkb  = lhalf * 8;   // bf16 units
        Frag bh, bl;
        bh.u[0] = *(const uint4*)&sB_hi[buf][bcol * LDSP + bkb];
        bh.u[1] = *(const uint4*)&sB_hi[buf][bcol * LDSP + bkb + 16];
        bl.u[0] = *(const uint4*)&sB_lo[buf][bcol * LDSP + bkb];
        bl.u[1] = *(const uint4*)&sB_lo[buf][bcol * LDSP + bkb + 16];
        acc[mi][ni] = __builtin_amdgcn_wmma_f32_16x16x32_bf16(
            false, ah.v, false, bh.v, (short)0, acc[mi][ni], false, false);
        acc[mi][ni] = __builtin_amdgcn_wmma_f32_16x16x32_bf16(
            false, ah.v, false, bl.v, (short)0, acc[mi][ni], false, false);
        acc[mi][ni] = __builtin_amdgcn_wmma_f32_16x16x32_bf16(
            false, al.v, false, bh.v, (short)0, acc[mi][ni], false, false);
      }
    }
  };

  // ---- pipelined main loop (2x unrolled ping-pong, NKC is even) ------------
  load_w_async(0, 0);
  load_x(0, 0);
  wait_async0();
  __syncthreads();

  for (int kc = 0; kc < NKC; kc += 2) {
    // stage kc+1 into buf1 while computing buf0
    load_w_async(kc + 1, 1);
    load_x(kc + 1, 1);
    compute(0);
    wait_async0();
    __syncthreads();

    // stage kc+2 into buf0 while computing buf1
    if (kc + 2 < NKC) {
      load_w_async(kc + 2, 0);
      load_x(kc + 2, 0);
    }
    compute(1);
    wait_async0();
    __syncthreads();
  }

  // ---- epilogue: C tile layout: VGPR r -> M = r + 8*(lane>=16), N = lane%16
#pragma unroll
  for (int mi = 0; mi < 2; ++mi) {
#pragma unroll
    for (int ni = 0; ni < 4; ++ni) {
      int row0 = bm0 + wave_m * 32 + mi * 16 + lhalf * 8;
      int col  = bn0 + wave_n * 64 + ni * 16 + l16;
      float bv = bias[col];
#pragma unroll
      for (int r = 0; r < 8; ++r)
        out[(size_t)(row0 + r) * GEMM_N + col] = acc[mi][ni][r] + bv;
    }
  }
}

// ---------------------------------------------------------------------------
extern "C" void kernel_launch(void* const* d_in, const int* in_sizes, int n_in,
                              void* d_out, int out_size, void* d_ws, size_t ws_size,
                              hipStream_t stream) {
  const float* x   = (const float*)d_in[0];   // [4, 4096, 768]
  const float* Af  = (const float*)d_in[1];   // [2048, 64]
  const float* Bf  = (const float*)d_in[2];   // [64, 1152]
  const float* Sp  = (const float*)d_in[3];   // [768, 3072]
  const float* bia = (const float*)d_in[4];   // [3072]
  float* out = (float*)d_out;                 // [16384, 3072]

  u16* wt_hi = (u16*)d_ws;                          // 768*3072 bf16 = 4.5 MiB
  u16* wt_lo = wt_hi + (size_t)GEMM_K * GEMM_N;     // + 4.5 MiB

  // Stage 1: 768*3072 elements / 256 threads = 9216 blocks
  build_wt_kernel<<<dim3((GEMM_K * GEMM_N) / 256), 256, 0, stream>>>(
      Af, Bf, Sp, wt_hi, wt_lo);

  // Stage 2: 128x128 tiles -> grid (3072/128, 16384/128) = (24, 128)
  kron_gemm_kernel<<<dim3(GEMM_N / 128, GEMM_M / 128), 256, 0, stream>>>(
      x, wt_hi, wt_lo, bia, out);
}
